// VQVAE_22565758173832
// MI455X (gfx1250) — matrix operand order repaired and checked by reading, not compiled, not run on hardware
//
#include <hip/hip_runtime.h>

// ---------------- problem dimensions ----------------
#define NTOK   4096
#define DMODEL 512
#define DHEAD  128
#define NHEAD  4
#define EDIM   512
#define FDIM   4096
#define KCODE  1024

#ifndef __has_builtin
#define __has_builtin(x) 0
#endif
#if __has_builtin(__builtin_amdgcn_global_load_async_to_lds_b128)
#define HAVE_ASYNC_B128 1
#else
#define HAVE_ASYNC_B128 0
#endif
#if __has_builtin(__builtin_amdgcn_s_wait_asynccnt)
#define HAVE_WAIT_ASYNC 1
#else
#define HAVE_WAIT_ASYNC 0
#endif

#if HAVE_ASYNC_B128
#if HAVE_WAIT_ASYNC
#define STAGE_WAIT(n) __builtin_amdgcn_s_wait_asynccnt(n)
#else
#define STAGE_WAIT(n) asm volatile("s_wait_asynccnt %0" :: "i"(n) : "memory")
#endif
#else
#define STAGE_WAIT(n) ((void)0)
#endif

// ---------------- WMMA types ----------------
typedef __attribute__((ext_vector_type(16))) __bf16 v16bf;
typedef __attribute__((ext_vector_type(8)))  float  v8f;
typedef int v4i_vec __attribute__((vector_size(16)));

union Frag16 { unsigned int u[8]; v16bf v; };

__device__ __forceinline__ unsigned short f2bf(float x) {
  unsigned int u = __float_as_uint(x);
  u += 0x7FFFu + ((u >> 16) & 1u);           // round to nearest even
  return (unsigned short)(u >> 16);
}

// 16-byte global -> LDS stage; async path (GLOBAL_LOAD_ASYNC_TO_LDS_B128) when available
__device__ __forceinline__ void stage16(const unsigned short* g, unsigned short* l) {
#if HAVE_ASYNC_B128
  __builtin_amdgcn_global_load_async_to_lds_b128(
      (__attribute__((address_space(1))) v4i_vec*)g,
      (__attribute__((address_space(3))) v4i_vec*)l, 0, 0);
#else
  *(uint4*)l = *(const uint4*)g;
#endif
}

// ---------------- tiled WMMA GEMM (bf16 in, f32 accum) ----------------
// C = act( scale * A[M,K] @ B^T + bias[N] ), A row-major [M,K] bf16,
// B stored row-major [N,K] bf16 (i.e. logical B(k,n) at B[n*ldb+k]).
// Block tile 128x128, 8 waves each computing 32x64 (2x4 WMMA tiles).
// Double-buffered LDS fed by async global->LDS copies (4 per wave per tile).
// flags: bit0 = relu; bits[2:1] = out mode (0: f32, 1: bf16, 2: bf16 transposed,
//        where mode 2 stores C^T[col*ldc + row]).
#define BM   128
#define BN   128
#define KC   32
#define LDSA 40
#define LDSB 40

__global__ __launch_bounds__(256) void wmma_gemm_bf16(
    const unsigned short* __restrict__ A, int lda,
    const unsigned short* __restrict__ B, int ldb,
    const float* __restrict__ bias,
    void* __restrict__ Cout, int ldc,
    int K, float scale, int flags)
{
  __shared__ __align__(16) unsigned short As[2][BM * LDSA];
  __shared__ __align__(16) unsigned short Bs[2][BN * LDSB];   // n-major: Bs[n][k]

  const int tid  = threadIdx.x;
  const int lane = tid & 31;
  const int wave = tid >> 5;            // 8 waves
  const int wm   = (wave >> 1) * 32;    // 4 wave-rows
  const int wn   = (wave & 1) * 64;     // 2 wave-cols
  const int bm0  = blockIdx.y * BM;
  const int bn0  = blockIdx.x * BN;

  const int l15 = lane & 15;
  const int hiA = (lane >= 16) ? 8  : 0;
  const int hiB = (lane >= 16) ? 8  : 0;

  // per-thread staging coordinates (128 rows x 4 chunks of 8 bf16, 2 passes)
  const int srow = tid >> 2;            // 0..63
  const int sch  = (tid & 3) * 8;       // 0,8,16,24

  v8f acc[2][4] = {};

  // stage(k0) -> buffer b: 2 A chunks + 2 B chunks per thread = 4 async ops/wave
  auto stage = [&](int b, int k0) {
    stage16(A + (size_t)(bm0 + srow)      * lda + k0 + sch, &As[b][(srow)      * LDSA + sch]);
    stage16(A + (size_t)(bm0 + srow + 64) * lda + k0 + sch, &As[b][(srow + 64) * LDSA + sch]);
    stage16(B + (size_t)(bn0 + srow)      * ldb + k0 + sch, &Bs[b][(srow)      * LDSB + sch]);
    stage16(B + (size_t)(bn0 + srow + 64) * ldb + k0 + sch, &Bs[b][(srow + 64) * LDSB + sch]);
  };

  stage(0, 0);
  int buf = 0;

  for (int k0 = 0; k0 < K; k0 += KC) {
    if (k0 + KC < K) {
      stage(buf ^ 1, k0 + KC);
      // speculative L2 prefetch of the tile after next (fault-safe per ISA 10.5)
      __builtin_prefetch(A + (size_t)(bm0 + srow) * lda + k0 + 2 * KC + sch, 0, 1);
      __builtin_prefetch(B + (size_t)(bn0 + srow) * ldb + k0 + 2 * KC + sch, 0, 1);
      STAGE_WAIT(4);      // previous tile's 4 copies done; new 4 still in flight
    } else {
      STAGE_WAIT(0);
    }
    __syncthreads();

    // ---- fragments per ISA VGPR layout ----
    Frag16 af[2], bf[4];
    #pragma unroll
    for (int mt = 0; mt < 2; ++mt) {
      const unsigned short* rowp = &As[buf][(wm + mt * 16 + l15) * LDSA];
      #pragma unroll
      for (int p = 0; p < 8; ++p) {
        const int kk = ((p < 4) ? (2 * p) : (16 + 2 * (p - 4))) + hiA;
        af[mt].u[p] = *(const unsigned int*)(rowp + kk);
      }
    }
    #pragma unroll
    for (int nt = 0; nt < 4; ++nt) {
      const unsigned short* rowp = &Bs[buf][(wn + nt * 16 + l15) * LDSB];
      #pragma unroll
      for (int p = 0; p < 8; ++p)
        bf[nt].u[p] = *(const unsigned int*)(rowp + ((p < 4) ? (2 * p) : (16 + 2 * (p - 4))) + hiB);
    }

    #pragma unroll
    for (int mt = 0; mt < 2; ++mt)
      #pragma unroll
      for (int nt = 0; nt < 4; ++nt)
        acc[mt][nt] = __builtin_amdgcn_wmma_f32_16x16x32_bf16(
            false, af[mt].v, false, bf[nt].v, (short)0, acc[mt][nt], false, false);

    __syncthreads();
    buf ^= 1;
  }

  // ---- epilogue ----
  const int relu    = flags & 1;
  const int outmode = flags >> 1;
  const int rhi     = (lane >= 16) ? 8 : 0;
  #pragma unroll
  for (int mt = 0; mt < 2; ++mt) {
    #pragma unroll
    for (int nt = 0; nt < 4; ++nt) {
      const int col = bn0 + wn + nt * 16 + l15;
      const float bb = bias ? bias[col] : 0.0f;
      #pragma unroll
      for (int r = 0; r < 8; ++r) {
        const int row = bm0 + wm + mt * 16 + r + rhi;
        float v = acc[mt][nt][r] * scale + bb;
        if (relu) v = fmaxf(v, 0.0f);
        if (outmode == 0)      ((float*)Cout)[(size_t)row * ldc + col] = v;
        else if (outmode == 1) ((unsigned short*)Cout)[(size_t)row * ldc + col] = f2bf(v);
        else                   ((unsigned short*)Cout)[(size_t)col * ldc + row] = f2bf(v);
      }
    }
  }
}

// ---------------- fp32 -> bf16 (flat, 4 elems/thread) ----------------
__global__ __launch_bounds__(256) void cvt_bf16(
    const float* __restrict__ src, unsigned short* __restrict__ dst)
{
  const size_t i = (size_t)blockIdx.x * 256 + threadIdx.x;
  const float4 f = ((const float4*)src)[i];
  unsigned long long pk = (unsigned long long)f2bf(f.x)
                        | ((unsigned long long)f2bf(f.y) << 16)
                        | ((unsigned long long)f2bf(f.z) << 32)
                        | ((unsigned long long)f2bf(f.w) << 48);
  *(unsigned long long*)(dst + i * 4) = pk;
}

// ---------------- fp32 [R,C] -> bf16 transposed [C,R] ----------------
__global__ __launch_bounds__(256) void cvt_bf16_t(
    const float* __restrict__ src, unsigned short* __restrict__ dst, int R, int C)
{
  __shared__ float t[32][33];
  const int c0 = blockIdx.x * 32, r0 = blockIdx.y * 32;
  const int lc = threadIdx.x & 31, wr = threadIdx.x >> 5;   // 32 x 8
  #pragma unroll
  for (int i = 0; i < 4; ++i)
    t[wr + 8 * i][lc] = src[(size_t)(r0 + wr + 8 * i) * C + c0 + lc];
  __syncthreads();
  #pragma unroll
  for (int i = 0; i < 4; ++i)
    dst[(size_t)(c0 + wr + 8 * i) * R + r0 + lc] = f2bf(t[lc][wr + 8 * i]);
}

// ---------------- row softmax: f32 in (scratch), bf16 out ----------------
__global__ __launch_bounds__(256) void softmax_rows(
    float* __restrict__ S, unsigned short* __restrict__ O, int ncols)
{
  float* p = S + (size_t)blockIdx.x * ncols;
  unsigned short* o = O + (size_t)blockIdx.x * ncols;
  const int tid = threadIdx.x;
  __shared__ float red[256];

  float mx = -3.402823466e+38f;
  for (int c = tid; c < ncols; c += 256) mx = fmaxf(mx, p[c]);
  red[tid] = mx; __syncthreads();
  for (int s = 128; s > 0; s >>= 1) { if (tid < s) red[tid] = fmaxf(red[tid], red[tid + s]); __syncthreads(); }
  mx = red[0]; __syncthreads();

  float sum = 0.0f;
  for (int c = tid; c < ncols; c += 256) { float e = __expf(p[c] - mx); p[c] = e; sum += e; }
  red[tid] = sum; __syncthreads();
  for (int s = 128; s > 0; s >>= 1) { if (tid < s) red[tid] += red[tid + s]; __syncthreads(); }
  const float inv = 1.0f / red[0];
  for (int c = tid; c < ncols; c += 256) o[c] = f2bf(p[c] * inv);
}

// ---------------- residual add + LayerNorm, f32 + bf16 outputs ----------------
__global__ __launch_bounds__(256) void add_layernorm(
    const float* __restrict__ x, const float* __restrict__ r,
    const float* __restrict__ g, const float* __restrict__ b,
    float* __restrict__ out, unsigned short* __restrict__ out_bf)
{
  const int row = blockIdx.x, tid = threadIdx.x;
  const float* px = x + (size_t)row * DMODEL;
  const float* pr = r + (size_t)row * DMODEL;
  const float v0 = px[tid] + pr[tid];
  const float v1 = px[tid + 256] + pr[tid + 256];

  __shared__ float red[256];
  red[tid] = v0 + v1; __syncthreads();
  for (int s = 128; s > 0; s >>= 1) { if (tid < s) red[tid] += red[tid + s]; __syncthreads(); }
  const float mu = red[0] * (1.0f / DMODEL); __syncthreads();

  const float d0 = v0 - mu, d1 = v1 - mu;
  red[tid] = d0 * d0 + d1 * d1; __syncthreads();
  for (int s = 128; s > 0; s >>= 1) { if (tid < s) red[tid] += red[tid + s]; __syncthreads(); }
  const float inv = rsqrtf(red[0] * (1.0f / DMODEL) + 1e-6f);

  const float y0 = d0 * inv * g[tid]       + b[tid];
  const float y1 = d1 * inv * g[tid + 256] + b[tid + 256];
  float* po = out + (size_t)row * DMODEL;
  po[tid] = y0; po[tid + 256] = y1;
  unsigned short* pb = out_bf + (size_t)row * DMODEL;
  pb[tid] = f2bf(y0); pb[tid + 256] = f2bf(y1);
}

// ---------------- codebook inverse norms ----------------
__global__ __launch_bounds__(256) void emb_invnorm(
    const float* __restrict__ emb, float* __restrict__ invn)
{
  const int row = blockIdx.x, tid = threadIdx.x;
  const float* p = emb + (size_t)row * EDIM;
  const float a = p[tid], b = p[tid + 256];
  __shared__ float red[256];
  red[tid] = a * a + b * b; __syncthreads();
  for (int s = 128; s > 0; s >>= 1) { if (tid < s) red[tid] += red[tid + s]; __syncthreads(); }
  if (tid == 0) invn[row] = rsqrtf(fmaxf(red[0], 1e-12f));
}

// ---------------- VQ: argmax cosine, one-hot + gather (f32 + bf16) ----------------
__global__ __launch_bounds__(256) void vq_select(
    const float* __restrict__ sims, const float* __restrict__ invn,
    const float* __restrict__ emb,
    float* __restrict__ onehot, float* __restrict__ vqfeat,
    unsigned short* __restrict__ vqfeat_bf)
{
  const int row = blockIdx.x, tid = threadIdx.x;
  const float* ps = sims + (size_t)row * KCODE;

  float best = -3.402823466e+38f;
  int   bi   = 0x7fffffff;
  for (int k = tid; k < KCODE; k += 256) {
    const float v = ps[k] * invn[k];
    if (v > best || (v == best && k < bi)) { best = v; bi = k; }
  }
  __shared__ float bv[256];
  __shared__ int   bx[256];
  bv[tid] = best; bx[tid] = bi; __syncthreads();
  for (int s = 128; s > 0; s >>= 1) {
    if (tid < s) {
      const float v2 = bv[tid + s]; const int i2 = bx[tid + s];
      if (v2 > bv[tid] || (v2 == bv[tid] && i2 < bx[tid])) { bv[tid] = v2; bx[tid] = i2; }
    }
    __syncthreads();
  }
  const int sel = bx[0];

  float* po = onehot + (size_t)row * KCODE;
  for (int k = tid; k < KCODE; k += 256) po[k] = (k == sel) ? 1.0f : 0.0f;
  const float* pe = emb + (size_t)sel * EDIM;
  float* pv = vqfeat + (size_t)row * EDIM;
  unsigned short* pb = vqfeat_bf + (size_t)row * EDIM;
  const float e0 = pe[tid], e1 = pe[tid + 256];
  pv[tid] = e0; pv[tid + 256] = e1;
  pb[tid] = f2bf(e0); pb[tid + 256] = f2bf(e1);
}

// ---------------- host-side orchestration ----------------
typedef unsigned short u16;

static inline void gemm(hipStream_t s,
                        const u16* A, int lda, const u16* B, int ldb,
                        const float* bias, void* C, int ldc,
                        int M, int Nc, int K, float scale, int outmode, bool relu)
{
  dim3 grid(Nc / BN, M / BM);
  wmma_gemm_bf16<<<grid, dim3(256), 0, s>>>(A, lda, B, ldb, bias, C, ldc, K, scale,
                                            (outmode << 1) | (relu ? 1 : 0));
}

#define MB ((size_t)1 << 20)
// f32 region
#define OF_SCORES (0 * MB)       // 64MB, SIMS overlays (16MB)
#define OF_SIMS   (0 * MB)
#define OF_T0     (96 * MB)      // 8MB
#define OF_OUT1   (104 * MB)     // 8MB
#define OF_X2F    (112 * MB)     // 8MB
#define OF_EMBINV (120 * MB)     // 4KB
// bf16 region
#define OF_ATTNBF (64 * MB)      // 32MB
#define OF_XBF    (121 * MB)     // 4MB each below
#define OF_QBF    (125 * MB)
#define OF_KBF    (129 * MB)
#define OF_VTBF   (133 * MB)     // v stored transposed [512][4096]
#define OF_CTXBF  (137 * MB)
#define OF_OUT1BF (141 * MB)
#define OF_FFNBF  (145 * MB)
#define OF_X2BF   (149 * MB)
#define OF_ENCBF  (153 * MB)
#define OF_WT     (157 * MB)     // 13 x 512x512 bf16 transposed weights (0.5MB each)
#define OF_FC2T   (164 * MB)     // 4MB  [4096][512]
#define OF_EMBBF  (168 * MB)     // 1MB
// total ~169MB

// output offsets (floats)
#define OUT_ENC ((size_t)0)
#define OUT_VQ  ((size_t)2097152)
#define OUT_IND ((size_t)4194304)
#define OUT_DEC ((size_t)8388608)
#define OUT_EMB ((size_t)25165824)

extern "C" void kernel_launch(void* const* d_in, const int* in_sizes, int n_in,
                              void* d_out, int out_size, void* d_ws, size_t ws_size,
                              hipStream_t stream)
{
  (void)in_sizes; (void)n_in; (void)out_size; (void)ws_size;
  auto P  = [&](int i) { return (const float*)d_in[i]; };
  char*  wsb = (char*)d_ws;
  auto F  = [&](size_t off) { return (float*)(wsb + off); };
  auto H  = [&](size_t off) { return (u16*)(wsb + off); };
  float* out = (float*)d_out;
  const float inv_sqrt_dh = 0.08838834764831843f;   // 1/sqrt(128)

  auto wt = [&](int i) { return (u16*)(wsb + OF_WT + (size_t)i * DMODEL * DMODEL * 2); };

  // ---- one-time (per call) weight conversions: W[K,N] -> bf16 W^T[N,K] ----
  for (int j = 0; j < 6; ++j)
    cvt_bf16_t<<<dim3(512 / 32, 512 / 32), 256, 0, stream>>>(P(1 + j),  wt(j),     512, 512);
  for (int j = 0; j < 6; ++j)
    cvt_bf16_t<<<dim3(512 / 32, 512 / 32), 256, 0, stream>>>(P(17 + j), wt(6 + j), 512, 512);
  cvt_bf16_t<<<dim3(512 / 32, 512 / 32), 256, 0, stream>>>(P(33), wt(12), 512, 512);         // fc1_w
  cvt_bf16_t<<<dim3(4096 / 32, 512 / 32), 256, 0, stream>>>(P(35), H(OF_FC2T), 512, 4096);   // fc2_w
  cvt_bf16<<<(KCODE * EDIM) / 1024, 256, 0, stream>>>(P(37), H(OF_EMBBF));                    // emb (already [N,K])
  cvt_bf16<<<(NTOK * DMODEL) / 1024, 256, 0, stream>>>(P(0), H(OF_XBF));                      // encoder input

  // transformer layer: fp32 x (residual), bf16 x (GEMM A); weight base b, transposed set wb
  auto run_layer = [&](const float* xf, const u16* xbf, int b, int wb) {
    gemm(stream, xbf, DMODEL, wt(wb + 0), DMODEL, P(b + 6), H(OF_QBF), DMODEL, NTOK, DMODEL, DMODEL, 1.0f, 1, false);
    gemm(stream, xbf, DMODEL, wt(wb + 1), DMODEL, P(b + 7), H(OF_KBF), DMODEL, NTOK, DMODEL, DMODEL, 1.0f, 1, false);
    gemm(stream, xbf, DMODEL, wt(wb + 2), DMODEL, P(b + 8), H(OF_VTBF), NTOK,   NTOK, DMODEL, DMODEL, 1.0f, 2, false);
    for (int h = 0; h < NHEAD; ++h) {
      gemm(stream, H(OF_QBF) + h * DHEAD, DMODEL, H(OF_KBF) + h * DHEAD, DMODEL,
           nullptr, F(OF_SCORES), NTOK, NTOK, NTOK, DHEAD, inv_sqrt_dh, 0, false);
      softmax_rows<<<NTOK, 256, 0, stream>>>(F(OF_SCORES), H(OF_ATTNBF), NTOK);
      gemm(stream, H(OF_ATTNBF), NTOK, H(OF_VTBF) + (size_t)h * DHEAD * NTOK, NTOK,
           nullptr, H(OF_CTXBF) + h * DHEAD, DMODEL, NTOK, DHEAD, NTOK, 1.0f, 1, false);
    }
    gemm(stream, H(OF_CTXBF), DMODEL, wt(wb + 3), DMODEL, P(b + 9), F(OF_T0), DMODEL, NTOK, DMODEL, DMODEL, 1.0f, 0, false);
    add_layernorm<<<NTOK, 256, 0, stream>>>(xf, F(OF_T0), P(b + 12), P(b + 13), F(OF_OUT1), H(OF_OUT1BF));
    gemm(stream, H(OF_OUT1BF), DMODEL, wt(wb + 4), DMODEL, P(b + 10), H(OF_FFNBF), DMODEL, NTOK, DMODEL, DMODEL, 1.0f, 1, true);
    gemm(stream, H(OF_FFNBF), DMODEL, wt(wb + 5), DMODEL, P(b + 11), F(OF_T0), DMODEL, NTOK, DMODEL, DMODEL, 1.0f, 0, false);
    add_layernorm<<<NTOK, 256, 0, stream>>>(F(OF_OUT1), F(OF_T0), P(b + 14), P(b + 15), F(OF_X2F), H(OF_X2BF));
  };

  // ---- encoder ----
  run_layer(P(0), H(OF_XBF), 1, 0);
  gemm(stream, H(OF_X2BF), DMODEL, wt(12), DMODEL, P(34), out + OUT_ENC, EDIM, NTOK, EDIM, DMODEL, 1.0f, 0, false);
  cvt_bf16<<<(NTOK * EDIM) / 1024, 256, 0, stream>>>(out + OUT_ENC, H(OF_ENCBF));

  // ---- VQ ----
  emb_invnorm<<<KCODE, 256, 0, stream>>>(P(37), F(OF_EMBINV));
  gemm(stream, H(OF_ENCBF), EDIM, H(OF_EMBBF), EDIM, nullptr, F(OF_SIMS), KCODE, NTOK, KCODE, EDIM, 1.0f, 0, false);
  vq_select<<<NTOK, 256, 0, stream>>>(F(OF_SIMS), F(OF_EMBINV), P(37),
                                      out + OUT_IND, out + OUT_VQ, H(OF_XBF));

  // ---- decoder ----
  run_layer(out + OUT_VQ, H(OF_XBF), 17, 6);
  gemm(stream, H(OF_X2BF), DMODEL, H(OF_FC2T), DMODEL, P(36), out + OUT_DEC, FDIM, NTOK, FDIM, DMODEL, 1.0f, 0, true);

  // ---- emb passthrough ----
  (void)hipMemcpyAsync(out + OUT_EMB, d_in[37], (size_t)KCODE * EDIM * sizeof(float),
                       hipMemcpyDeviceToDevice, stream);
}
